// RGCNClassifier_88648124990510
// MI455X (gfx1250) — matrix operand order repaired and checked by reading, not compiled, not run on hardware
//
#include <hip/hip_runtime.h>

typedef __attribute__((ext_vector_type(16))) __bf16 v16bf;
typedef __attribute__((ext_vector_type(8)))  float  v8f;

#define NREL 3
#define HDIM 96

__device__ __forceinline__ unsigned short f2bf(float f) {
  unsigned u = __float_as_uint(f);
  unsigned r = u + 0x7FFFu + ((u >> 16) & 1u);   // round-to-nearest-even
  return (unsigned short)(r >> 16);
}

__device__ __forceinline__ void atomAddF(float* p, float v) {
  __hip_atomic_fetch_add(p, v, __ATOMIC_RELAXED, __HIP_MEMORY_SCOPE_AGENT);
}

// ---------------- embedding gather: x1[n][d] = emb[tokens[n]][d] -------------
__global__ void embed_kernel(const float* __restrict__ emb, const int* __restrict__ tok,
                             float* __restrict__ x, int N, int D) {
  long long id = (long long)blockIdx.x * blockDim.x + threadIdx.x;
  if (id >= (long long)N * D) return;
  int n = (int)(id / D), d = (int)(id % D);
  x[(size_t)n * D + d] = emb[(size_t)tok[n] * D + d];
}

// ---------------- per-(node,relation) edge counts ----------------------------
__global__ void count_kernel(const int* __restrict__ dst, const int* __restrict__ et,
                             int* __restrict__ cnt, int E) {
  int e = blockIdx.x * blockDim.x + threadIdx.x;
  if (e >= E) return;
  atomicAdd(&cnt[dst[e] * NREL + et[e]], 1);
}

__global__ void inv_kernel(const int* __restrict__ cnt, float* __restrict__ inv, int n) {
  int i = blockIdx.x * blockDim.x + threadIdx.x;
  if (i >= n) return;
  inv[i] = 1.0f / fmaxf((float)cnt[i], 1.0f);
}

// ---------------- edge aggregation: sums[dst,et,:] += x[src,:] ---------------
template <int DIN>
__global__ void agg_kernel(const float* __restrict__ x, const int* __restrict__ src,
                           const int* __restrict__ dst, const int* __restrict__ et,
                           float* __restrict__ sums, int E) {
  constexpr int CH = DIN / 4;
  long long id = (long long)blockIdx.x * blockDim.x + threadIdx.x;
  if (id >= (long long)E * CH) return;
  int e = (int)(id / CH), c = (int)(id % CH);
  int s = src[e], d = dst[e], r = et[e];
  const float4 v = *(const float4*)&x[(size_t)s * DIN + c * 4];
  float* p = &sums[((size_t)d * NREL + r) * DIN + c * 4];
  atomAddF(p + 0, v.x); atomAddF(p + 1, v.y);
  atomAddF(p + 2, v.z); atomAddF(p + 3, v.w);
}

// ------------- pre-swizzle [Wcat = W(3*DIN) ++ root(DIN)] into B-fragments ---
// B frag layout used here (contiguous k, same as A): lane<16: elem e = B[k0+e][nt*16+lane]
//                                           lane>=16: elem e = B[k0+16+e][...]
template <int DIN>
__global__ void prep_w_kernel(const float* __restrict__ W, const float* __restrict__ root,
                              unsigned short* __restrict__ wfrag) {
  constexpr int KC = 4 * DIN;
  int id = blockIdx.x * blockDim.x + threadIdx.x;
  if (id >= KC * HDIM) return;
  int k = id / HDIM, h = id % HDIM;
  float v = (k < NREL * DIN) ? W[(size_t)k * HDIM + h]
                             : root[(size_t)(k - NREL * DIN) * HDIM + h];
  int kt = k >> 5, ko = k & 31;
  int nt = h >> 4, ln = h & 15;
  int lane, e;
  if (ko < 16) { lane = ln;      e = ko;      }
  else         { lane = ln + 16; e = ko - 16; }
  wfrag[(((size_t)kt * 6 + nt) * 32 + lane) * 16 + e] = f2bf(v);
}

// ---------------- fused mean + GEMM(+root) + bias + ReLU via WMMA ------------
// out[16-node tile][96] = bf16([mean_r0|mean_r1|mean_r2|x]) @ bf16(Wcat) (+b,relu)
template <int DIN>
__global__ __launch_bounds__(128) void transform_kernel(
    const float* __restrict__ sums, const float* __restrict__ inv,
    const float* __restrict__ xin, const unsigned short* __restrict__ wfrag,
    const float* __restrict__ bias, float* __restrict__ out, int N, int ntiles) {
  constexpr int KC = 4 * DIN;        // 256 (L1) / 384 (L2)
  constexpr int KT = KC / 32;        // k-tiles
  constexpr int NT = HDIM / 16;      // 6 output column tiles
  constexpr int BFRAG = KT * NT * 512;  // ushorts for all B fragments
  constexpr int AFRAG = KT * 512;       // ushorts per wave A panel

  extern __shared__ uint4 lds4[];
  unsigned short* sB = (unsigned short*)lds4;
  const int wave = threadIdx.x >> 5;
  const int lane = threadIdx.x & 31;
  unsigned short* sA = sB + BFRAG + wave * AFRAG;

  // block-cooperative: all B fragments -> LDS once
  {
    const uint4* g = (const uint4*)wfrag;
    uint4* d = lds4;
    for (int i = threadIdx.x; i < BFRAG / 8; i += blockDim.x) d[i] = g[i];
  }
  __syncthreads();

  const int nwaves = gridDim.x * (blockDim.x >> 5);
  for (int tile = blockIdx.x * (blockDim.x >> 5) + wave; tile < ntiles; tile += nwaves) {
    const int node0 = tile * 16;

    // stage A panel (coalesced global reads, write in A-fragment order)
    for (int idx = lane; idx < 16 * KC; idx += 32) {
      const int m = idx / KC;
      const int k = idx % KC;
      const int node = node0 + m;
      float v = 0.0f;
      if (node < N) {
        if (k < NREL * DIN)
          v = sums[(size_t)node * (NREL * DIN) + k] * inv[node * NREL + k / DIN];
        else
          v = xin[(size_t)node * DIN + (k - NREL * DIN)];
      }
      const int kt = k >> 5, ko = k & 31, g = ko >> 3;
      const int ln = (g & 2) ? (m + 16) : m;
      const int e  = ko & 15;
      sA[((size_t)kt * 32 + ln) * 16 + e] = f2bf(v);
    }
    asm volatile("s_wait_dscnt 0" ::: "memory");  // wave-internal LDS visibility

    v8f acc[NT] = {};
#pragma unroll
    for (int kt = 0; kt < KT; ++kt) {
      v16bf a = *(const v16bf*)&sA[kt * 32 * 16 + lane * 16];
#pragma unroll
      for (int nt = 0; nt < NT; ++nt) {
        v16bf b = *(const v16bf*)&sB[((kt * NT + nt) * 32 + lane) * 16];
        acc[nt] = __builtin_amdgcn_wmma_f32_16x16x32_bf16(
            false, a, false, b, (short)0, acc[nt], false, false);
      }
    }

    // C/D layout: row M = (lane>>4)*8 + r, col N = nt*16 + (lane&15)
    const int col0 = lane & 15;
    const int mbase = (lane >> 4) * 8;
#pragma unroll
    for (int nt = 0; nt < NT; ++nt) {
      const int col = nt * 16 + col0;
      const float bv = bias[col];
#pragma unroll
      for (int r = 0; r < 8; ++r) {
        int node = node0 + mbase + r;
        if (node < N) {
          float v = acc[nt][r] + bv;
          out[(size_t)node * HDIM + col] = v > 0.0f ? v : 0.0f;
        }
      }
    }
  }
}

// ---------------- graph mean pool + linear head ------------------------------
__global__ void pool_feat_kernel(const float* __restrict__ h, const int* __restrict__ batch,
                                 float* __restrict__ pooled, int N) {
  long long id = (long long)blockIdx.x * blockDim.x + threadIdx.x;
  if (id >= (long long)N * HDIM) return;
  int n = (int)(id / HDIM), c = (int)(id % HDIM);
  atomAddF(&pooled[(size_t)batch[n] * HDIM + c], h[(size_t)n * HDIM + c]);
}

__global__ void pool_cnt_kernel(const int* __restrict__ batch, int* __restrict__ gcnt, int N) {
  int n = blockIdx.x * blockDim.x + threadIdx.x;
  if (n >= N) return;
  atomicAdd(&gcnt[batch[n]], 1);
}

__global__ void head_kernel(const float* __restrict__ pooled, const int* __restrict__ gcnt,
                            const float* __restrict__ linW, const float* __restrict__ linb,
                            float* __restrict__ out, int G) {
  int id = blockIdx.x * blockDim.x + threadIdx.x;
  if (id >= G * 8) return;
  int g = id / 8, c = id % 8;
  float invc = 1.0f / fmaxf((float)gcnt[g], 1.0f);
  float acc = 0.0f;
  const float* pr = &pooled[(size_t)g * HDIM];
#pragma unroll 8
  for (int h = 0; h < HDIM; ++h) acc += pr[h] * linW[h * 8 + c];
  out[(size_t)g * 8 + c] = acc * invc + linb[c];
}

// ---------------------------------------------------------------------------
static inline int gridFor(long long n, int b) { return (int)((n + b - 1) / b); }

extern "C" void kernel_launch(void* const* d_in, const int* in_sizes, int n_in,
                              void* d_out, int out_size, void* d_ws, size_t ws_size,
                              hipStream_t stream) {
  const int*   tokens = (const int*)d_in[0];
  const int*   eidx   = (const int*)d_in[1];
  const int*   etype  = (const int*)d_in[2];
  const int*   batch  = (const int*)d_in[3];
  const float* emb    = (const float*)d_in[5];
  const float* W1     = (const float*)d_in[6];
  const float* root1  = (const float*)d_in[7];
  const float* b1     = (const float*)d_in[8];
  const float* W2     = (const float*)d_in[9];
  const float* root2  = (const float*)d_in[10];
  const float* b2     = (const float*)d_in[11];
  const float* linW   = (const float*)d_in[12];
  const float* linb   = (const float*)d_in[13];

  const int N = in_sizes[0];
  const int E = in_sizes[2];
  const int G = out_size / 8;
  const int D1 = 64;
  const int* src = eidx;
  const int* dst = eidx + E;

  // -------- workspace layout (x1 region is aliased by h2 after x1 dies) -----
  char* ws = (char*)d_ws;
  size_t off = 0;
  auto take = [&](size_t bytes) -> char* {
    char* p = ws + off;
    off += (bytes + 255) & ~(size_t)255;
    return p;
  };
  float* xh2   = (float*)take((size_t)N * HDIM * 4);       // x1 (stride 64), later h2
  float* h1    = (float*)take((size_t)N * HDIM * 4);
  float* sums  = (float*)take((size_t)N * NREL * HDIM * 4);
  int*   cnt   = (int*)  take((size_t)N * NREL * 4);
  float* inv   = (float*)take((size_t)N * NREL * 4);
  float* pooled= (float*)take((size_t)G * HDIM * 4);
  int*   gcnt  = (int*)  take((size_t)G * 4);
  unsigned short* wf1 = (unsigned short*)take((size_t)4 * D1   * HDIM * 2);
  unsigned short* wf2 = (unsigned short*)take((size_t)4 * HDIM * HDIM * 2);
  (void)ws_size; (void)n_in;

  float* x1 = xh2;  // [N,64]
  float* h2 = xh2;  // [N,96] (after layer-1 transform, x1 is dead)

  const int B = 256;
  const int ntiles = (N + 15) / 16;
  const size_t lds1 = (size_t)(8  * 6 * 512 + 4 * 8  * 512) * 2;  // 80 KB
  const size_t lds2 = (size_t)(12 * 6 * 512 + 4 * 12 * 512) * 2;  // 120 KB

  hipMemsetAsync(cnt,    0, (size_t)N * NREL * 4, stream);
  hipMemsetAsync(pooled, 0, (size_t)G * HDIM * 4, stream);
  hipMemsetAsync(gcnt,   0, (size_t)G * 4, stream);
  hipMemsetAsync(sums,   0, (size_t)N * NREL * D1 * 4, stream);

  embed_kernel<<<gridFor((long long)N * D1, B), B, 0, stream>>>(emb, tokens, x1, N, D1);
  count_kernel<<<gridFor(E, B), B, 0, stream>>>(dst, etype, cnt, E);
  inv_kernel<<<gridFor((long long)N * NREL, B), B, 0, stream>>>(cnt, inv, N * NREL);
  prep_w_kernel<64><<<gridFor(256 * HDIM, B), B, 0, stream>>>(W1, root1, wf1);
  prep_w_kernel<96><<<gridFor(384 * HDIM, B), B, 0, stream>>>(W2, root2, wf2);

  // ---- layer 1 ----
  agg_kernel<64><<<gridFor((long long)E * 16, B), B, 0, stream>>>(x1, src, dst, etype, sums, E);
  transform_kernel<64><<<1024, 128, lds1, stream>>>(sums, inv, x1, wf1, b1, h1, N, ntiles);

  // ---- layer 2 ----
  hipMemsetAsync(sums, 0, (size_t)N * NREL * HDIM * 4, stream);
  agg_kernel<96><<<gridFor((long long)E * 24, B), B, 0, stream>>>(h1, src, dst, etype, sums, E);
  transform_kernel<96><<<1024, 128, lds2, stream>>>(sums, inv, h1, wf2, b2, h2, N, ntiles);

  // ---- pool + head ----
  pool_feat_kernel<<<gridFor((long long)N * HDIM, B), B, 0, stream>>>(h2, batch, pooled, N);
  pool_cnt_kernel<<<gridFor(N, B), B, 0, stream>>>(batch, gcnt, N);
  head_kernel<<<gridFor((long long)G * 8, B), B, 0, stream>>>(pooled, gcnt, linW, linb,
                                                              (float*)d_out, G);
}
